// SGCA_14276471292399
// MI455X (gfx1250) — compile-verified, matrix operand order checked by reading, not run on hardware
//
#include <hip/hip_runtime.h>
#include <hip/hip_bf16.h>

// ---------------------------------------------------------------------------
// CDNA5 / gfx1250 implementation of the SGConv+ARMA GNN pipeline.
//  * Dense GEMMs: v_wmma_f32_16x16x32_f16 (wave32, 16x16 tile per wave),
//    A-tiles double-buffered in LDS via the CDNA5 async copy engine
//    (global_load_async_to_lds_b128, pipelined with s_wait_asynccnt 0x8).
//  * Sparse propagation: wave-per-edge gather + global_atomic_add_f32 scatter
//    (the bandwidth-bound part of the workload at ~5-6 GB of edge traffic).
// ---------------------------------------------------------------------------

typedef __attribute__((ext_vector_type(16))) _Float16 v16h;
typedef __attribute__((ext_vector_type(8)))  float    v8f;

#define WAVE 32

// 16-byte async copy global -> LDS (ASYNCcnt-tracked, no VGPR round trip).
// Generic LDS pointer's low 32 bits equal the LDS byte offset (ISA 10.2).
__device__ __forceinline__ void async_copy16(const float* gsrc, float* ldst) {
    unsigned lofs = (unsigned)(uintptr_t)ldst;
    unsigned long long gaddr = (unsigned long long)(uintptr_t)gsrc;
    asm volatile("global_load_async_to_lds_b128 %0, %1, off"
                 :: "v"(lofs), "v"(gaddr) : "memory");
}
// Wait for all but the newest 8 async ops (previous chunk resident; async
// loads complete in order, so the <=8 outstanding are the newest chunk).
__device__ __forceinline__ void async_wait8() {
    asm volatile("s_wait_asynccnt 0x8" ::: "memory");
}
__device__ __forceinline__ void async_wait0() {
    asm volatile("s_wait_asynccnt 0x0" ::: "memory");
}

// ---------------------------------------------------------------------------
// WMMA GEMM:  C[M x Dout] = op( A[M x K] * W[K x Dout] (+ bias) (+ C) )
// Block: 256 threads = 8 waves. Block tile: 128 rows x 16 cols, K chunk 64
// (two v_wmma_f32_16x16x32_f16 per chunk). A staged fp32 in LDS via async
// copy, double-buffered and software-pipelined. Out-of-range A segments are
// address-clamped (not zeroed): a k-column only contributes A[m][k]*B[k][n],
// and B is zero-staged for k>=K / n>=Dout, so clamped A garbage is masked.
// ---------------------------------------------------------------------------
template <bool ADD_C, bool RELU, bool BIAS>
__global__ __launch_bounds__(256)
void wmma_gemm_kernel(const float* __restrict__ A,
                      const float* __restrict__ W,
                      const float* __restrict__ bias,
                      float* __restrict__ C,
                      int M, int K, int Dout) {
    const int BM = 128, BK = 64;
    __shared__ float    sA[2][BM * BK];   // fp32 A chunks, 2 x 32 KB
    __shared__ _Float16 sB[2][16 * BK];   // f16 W chunks transposed [n][k], 2 x 2 KB

    const int tid  = threadIdx.x;
    const int wave = tid >> 5;
    const int lane = tid & 31;
    const int rowBase = blockIdx.x * BM;
    const int colBase = blockIdx.y * 16;
    const int nK = (K + BK - 1) / BK;
    const bool vec_ok = ((K & 3) == 0);   // 16B-aligned rows (true for all dims here)

    // Stage A chunk kc into buffer buf: 2048 x 16B async segments.
    auto stage_A = [&](int kc, int buf) {
        const int k0 = kc * BK;
        if (vec_ok) {
            const int kvalid = (K - k0 < BK) ? (K - k0) : BK;  // multiple of 4
#pragma unroll
            for (int j = 0; j < 8; ++j) {
                const int seg = tid + j * 256;     // 16 segments per row
                const int r   = seg >> 4;
                const int kq  = (seg & 15) * 4;
                // Clamp OOB segments to a valid address (masked by zero B).
                int gr = rowBase + r; if (gr > M - 1) gr = M - 1;
                const int kqq = (kq < kvalid) ? kq : 0;
                async_copy16(A + (size_t)gr * K + k0 + kqq, &sA[buf][r * BK + kq]);
            }
        } else {
            for (int i = tid; i < BM * BK; i += 256) {
                const int r = i >> 6, k = i & 63;
                const int gr = rowBase + r, gk = k0 + k;
                sA[buf][r * BK + k] = (gr < M && gk < K) ? A[(size_t)gr * K + gk] : 0.0f;
            }
        }
    };

    // Stage W chunk kc transposed into buffer buf (zero-padded: this is what
    // masks the clamped A segments). Load-4-then-store-4 so loads pipeline.
    auto stage_B = [&](int kc, int buf) {
        const int k0 = kc * BK;
        float v[4];
#pragma unroll
        for (int j = 0; j < 4; ++j) {
            const int i = tid + j * 256;
            const int n = i & 15, k = i >> 4;    // consecutive tid -> consecutive gn
            const int gn = colBase + n, gk = k0 + k;
            v[j] = (gn < Dout && gk < K) ? W[(size_t)gk * Dout + gn] : 0.0f;
        }
#pragma unroll
        for (int j = 0; j < 4; ++j) {
            const int i = tid + j * 256;
            const int n = i & 15, k = i >> 4;
            sB[buf][n * BK + k] = (_Float16)v[j];
        }
    };

    v8f acc = {};

    // Software pipeline: prefetch chunk kc+1 while computing chunk kc.
    stage_A(0, 0);
    stage_B(0, 0);
    for (int kc = 0; kc < nK; ++kc) {
        const int cur = kc & 1, nxt = cur ^ 1;
        if (kc + 1 < nK) {
            stage_A(kc + 1, nxt);
            stage_B(kc + 1, nxt);
            async_wait8();               // chunk kc resident; kc+1 in flight
        } else {
            async_wait0();
        }
        __syncthreads();

        // Fragments per the CDNA5 16-bit WMMA VGPR layout:
        //  lane&15 selects m (A) / n (B); (lane>>4)*8 selects the K phase.
        //  v16h elems 0..7  -> K = kb+koff+0..7
        //  v16h elems 8..15 -> K = kb+16+koff+0..7
        const int mn   = lane & 15;
        const int koff = (lane >> 4) * 8;
        const float*    ap = &sA[cur][(wave * 16 + mn) * BK];
        const _Float16* bp = &sB[cur][mn * BK];
#pragma unroll
        for (int s = 0; s < 2; ++s) {
            const int kb = s * 32;
            v16h af, bf;
#pragma unroll
            for (int j = 0; j < 8; ++j) {
                af[j]     = (_Float16)ap[kb + koff + j];
                af[8 + j] = (_Float16)ap[kb + 16 + koff + j];
                bf[j]     = bp[kb + koff + j];
                bf[8 + j] = bp[kb + 16 + koff + j];
            }
            acc = __builtin_amdgcn_wmma_f32_16x16x32_f16(
                /*neg_a=*/false, af, /*neg_b=*/false, bf,
                /*c_mod=*/(short)0, acc, /*reuse_a=*/false, /*reuse_b=*/false);
        }
        __syncthreads();
    }

    // Write back: acc[r] holds (m = r + (lane>>4)*8, n = lane&15) of the tile.
    const int n     = colBase + (lane & 15);
    const int rbase = rowBase + wave * 16 + ((lane >> 4) * 8);
#pragma unroll
    for (int r = 0; r < 8; ++r) {
        const int row = rbase + r;
        if (row < M && n < Dout) {
            float v = acc[r];
            if (BIAS)  v += bias[n];
            if (ADD_C) v += C[(size_t)row * Dout + n];
            if (RELU)  v = fmaxf(v, 0.0f);
            C[(size_t)row * Dout + n] = v;
        }
    }
}

// ---------------------------------------------------------------------------
// Sparse graph kernels
// ---------------------------------------------------------------------------
__global__ void degree_kernel(const int* __restrict__ dst, float* __restrict__ deg, int E) {
    int e = blockIdx.x * blockDim.x + threadIdx.x;
    if (e < E) atomicAdd(&deg[dst[e]], 1.0f);
}

__global__ void dinv_kernel(const float* __restrict__ deg,
                            float* __restrict__ dinv_sl,
                            float* __restrict__ dinv_ns, int N) {
    int i = blockIdx.x * blockDim.x + threadIdx.x;
    if (i < N) {
        float d = deg[i];
        dinv_sl[i] = rsqrtf(d + 1.0f);                    // self-loop norm (deg+1 >= 1)
        dinv_ns[i] = (d > 0.0f) ? rsqrtf(d) : 0.0f;       // no-self-loop norm
    }
}

// One wave per edge; lanes stride the feature row (coalesced load + f32 atomic).
__global__ __launch_bounds__(256)
void edge_prop_kernel(const float* __restrict__ x,
                      const int* __restrict__ src, const int* __restrict__ dst,
                      const float* __restrict__ dinv,
                      float* __restrict__ agg, int D, int E) {
    int e = blockIdx.x * (blockDim.x / WAVE) + (threadIdx.x >> 5);
    if (e >= E) return;
    const int lane = threadIdx.x & 31;
    const int s = src[e], d = dst[e];
    const float w = dinv[s] * dinv[d];
    const float* xr = x + (size_t)s * D;
    float* ar = agg + (size_t)d * D;
    for (int f = lane; f < D; f += WAVE)
        atomicAdd(&ar[f], xr[f] * w);
}

__global__ void selfloop_kernel(const float* __restrict__ x,
                                const float* __restrict__ dinv_sl,
                                float* __restrict__ agg, int D, int N) {
    size_t idx = (size_t)blockIdx.x * blockDim.x + threadIdx.x;
    if (idx >= (size_t)N * D) return;
    int i = (int)(idx / D);
    float di = dinv_sl[i];
    agg[idx] += x[idx] * (di * di);
}

// ---------------------------------------------------------------------------
// BatchNorm (training-mode batch stats, biased variance)
// ---------------------------------------------------------------------------
__global__ __launch_bounds__(256)
void bn_stats_kernel(const float* __restrict__ h, float* __restrict__ mean,
                     float* __restrict__ var, int N, int D) {
    __shared__ float ss[256], ss2[256];
    const int f = blockIdx.x;
    float s = 0.0f, s2 = 0.0f;
    for (int i = threadIdx.x; i < N; i += 256) {
        float v = h[(size_t)i * D + f];
        s += v; s2 += v * v;
    }
    ss[threadIdx.x] = s; ss2[threadIdx.x] = s2;
    __syncthreads();
    for (int st = 128; st > 0; st >>= 1) {
        if (threadIdx.x < st) {
            ss[threadIdx.x]  += ss[threadIdx.x + st];
            ss2[threadIdx.x] += ss2[threadIdx.x + st];
        }
        __syncthreads();
    }
    if (threadIdx.x == 0) {
        float mu = ss[0] / (float)N;
        mean[f] = mu;
        var[f]  = ss2[0] / (float)N - mu * mu;
    }
}

__global__ void bn_apply_kernel(float* __restrict__ h,
                                const float* __restrict__ mean,
                                const float* __restrict__ var,
                                const float* __restrict__ gamma,
                                const float* __restrict__ beta,
                                int N, int D, float eps) {
    size_t idx = (size_t)blockIdx.x * blockDim.x + threadIdx.x;
    if (idx >= (size_t)N * D) return;
    int f = (int)(idx % D);
    h[idx] = (h[idx] - mean[f]) * rsqrtf(var[f] + eps) * gamma[f] + beta[f];
}

// ---------------------------------------------------------------------------
// global_add_pool: scatter node features into per-graph accumulators
// ---------------------------------------------------------------------------
__global__ void pool_kernel(const float* __restrict__ a, const int* __restrict__ batch,
                            float* __restrict__ g, int N, int D) {
    size_t idx = (size_t)blockIdx.x * blockDim.x + threadIdx.x;
    if (idx >= (size_t)N * D) return;
    int i = (int)(idx / D), f = (int)(idx % D);
    atomicAdd(&g[(size_t)batch[i] * D + f], a[idx]);
}

// ---------------------------------------------------------------------------
// Host orchestration
// ---------------------------------------------------------------------------
static inline dim3 gemm_grid(int M, int Dout) {
    return dim3((M + 127) / 128, (Dout + 15) / 16);
}

extern "C" void kernel_launch(void* const* d_in, const int* in_sizes, int n_in,
                              void* d_out, int out_size, void* d_ws, size_t ws_size,
                              hipStream_t stream) {
    const float* x     = (const float*)d_in[0];
    const int*   src   = (const int*)  d_in[1];
    const int*   dst   = (const int*)  d_in[2];
    const int*   batch = (const int*)  d_in[3];
    const float* W1 = (const float*)d_in[4];  const float* b1 = (const float*)d_in[5];
    const float* W2 = (const float*)d_in[6];  const float* b2 = (const float*)d_in[7];
    const float* gamma = (const float*)d_in[8];
    const float* beta  = (const float*)d_in[9];
    const float* Wa_init = (const float*)d_in[10];
    const float* Wa_root = (const float*)d_in[11];
    const float* ba      = (const float*)d_in[12];
    const float* Wf1 = (const float*)d_in[13]; const float* bf1 = (const float*)d_in[14];
    const float* Wf2 = (const float*)d_in[15]; const float* bf2 = (const float*)d_in[16];
    const float* Wf3 = (const float*)d_in[17]; const float* bf3 = (const float*)d_in[18];
    const float* Wf4 = (const float*)d_in[19]; const float* bf4 = (const float*)d_in[20];
    float* out = (float*)d_out;

    // Derive dimensions from input sizes (dim=100, D0=128, H1=200, H2=300).
    const int D   = in_sizes[5];            // dim
    const int D0  = in_sizes[4] / D;        // input feature width (128)
    const int N   = in_sizes[0] / D0;       // nodes
    const int E   = in_sizes[1];            // edges
    const int H1  = in_sizes[14];           // 2*dim
    const int H2  = in_sizes[16];           // 3*dim
    const int NG  = out_size;               // graphs (out = [NG,1])
    const float EPS = 1e-5f;

    // Workspace layout (floats), with reuse.
    float* ws = (float*)d_ws;
    size_t off = 0;
    float* deg  = ws + off; off += N;
    float* dsl  = ws + off; off += N;       // self-loop dinv
    float* dns  = ws + off; off += N;       // no-self-loop dinv
    float* BA   = ws + off; off += (size_t)N * D0;  // agg1 / agg2 / aggA / a
    float* BH1  = ws + off; off += (size_t)N * D;   // h1, then t = h2 @ Wa_init
    float* BH2  = ws + off; off += (size_t)N * D;   // h2 (post-BN)
    float* mean = ws + off; off += D;
    float* var  = ws + off; off += D;
    float* G    = ws + off; off += (size_t)NG * D;
    float* M1   = ws + off; off += (size_t)NG * H1;
    float* M2   = ws + off; off += (size_t)NG * H2;
    float* M3   = ws + off; off += (size_t)NG * H1;
    if (off * sizeof(float) > ws_size) return;  // insufficient scratch

    const int TB = 256;
    const size_t ND0 = (size_t)N * D0, ND = (size_t)N * D;

    // --- degrees + normalizers -------------------------------------------
    hipMemsetAsync(deg, 0, (size_t)N * sizeof(float), stream);
    hipMemsetAsync(BA,  0, ND0 * sizeof(float), stream);
    degree_kernel<<<(E + TB - 1) / TB, TB, 0, stream>>>(dst, deg, E);
    dinv_kernel<<<(N + TB - 1) / TB, TB, 0, stream>>>(deg, dsl, dns, N);

    // --- SGConv 1: h1 = ReLU(prop_sl(x) @ W1 + b1) -----------------------
    edge_prop_kernel<<<(E + 7) / 8, TB, 0, stream>>>(x, src, dst, dsl, BA, D0, E);
    selfloop_kernel<<<(unsigned)((ND0 + TB - 1) / TB), TB, 0, stream>>>(x, dsl, BA, D0, N);
    wmma_gemm_kernel<false, true, true><<<gemm_grid(N, D), TB, 0, stream>>>(
        BA, W1, b1, BH1, N, D0, D);

    // --- SGConv 2: h2 = ReLU(prop_sl(h1) @ W2 + b2) ----------------------
    hipMemsetAsync(BA, 0, ND * sizeof(float), stream);
    edge_prop_kernel<<<(E + 7) / 8, TB, 0, stream>>>(BH1, src, dst, dsl, BA, D, E);
    selfloop_kernel<<<(unsigned)((ND + TB - 1) / TB), TB, 0, stream>>>(BH1, dsl, BA, D, N);
    wmma_gemm_kernel<false, true, true><<<gemm_grid(N, D), TB, 0, stream>>>(
        BA, W2, b2, BH2, N, D, D);

    // --- BatchNorm (in place on h2) --------------------------------------
    bn_stats_kernel<<<D, TB, 0, stream>>>(BH2, mean, var, N, D);
    bn_apply_kernel<<<(unsigned)((ND + TB - 1) / TB), TB, 0, stream>>>(
        BH2, mean, var, gamma, beta, N, D, EPS);

    // --- ARMAConv: a = ReLU(prop_ns(h2 @ Wa_init) + h2 @ Wa_root + ba) ---
    wmma_gemm_kernel<false, false, false><<<gemm_grid(N, D), TB, 0, stream>>>(
        BH2, Wa_init, nullptr, BH1, N, D, D);                 // t -> BH1
    hipMemsetAsync(BA, 0, ND * sizeof(float), stream);
    edge_prop_kernel<<<(E + 7) / 8, TB, 0, stream>>>(BH1, src, dst, dns, BA, D, E);
    wmma_gemm_kernel<true, true, true><<<gemm_grid(N, D), TB, 0, stream>>>(
        BH2, Wa_root, ba, BA, N, D, D);                       // BA = ReLU(BA + h2 Wroot + ba)

    // --- global_add_pool --------------------------------------------------
    hipMemsetAsync(G, 0, (size_t)NG * D * sizeof(float), stream);
    pool_kernel<<<(unsigned)((ND + TB - 1) / TB), TB, 0, stream>>>(BA, batch, G, N, D);

    // --- MLP head ---------------------------------------------------------
    wmma_gemm_kernel<false, true, true><<<gemm_grid(NG, H1), TB, 0, stream>>>(
        G, Wf1, bf1, M1, NG, D, H1);
    wmma_gemm_kernel<false, true, true><<<gemm_grid(NG, H2), TB, 0, stream>>>(
        M1, Wf2, bf2, M2, NG, H1, H2);
    wmma_gemm_kernel<false, true, true><<<gemm_grid(NG, H1), TB, 0, stream>>>(
        M2, Wf3, bf3, M3, NG, H2, H1);
    wmma_gemm_kernel<false, false, true><<<gemm_grid(NG, 1), TB, 0, stream>>>(
        M3, Wf4, bf4, out, NG, H1, 1);
}
